// Attention_5720896438407
// MI455X (gfx1250) — compile-verified, matrix operand order checked by reading, not run on hardware
//
#include <hip/hip_runtime.h>

// Problem constants (reference): B=2, S=4096, HID=2048, H=16, D=128, KV=1, KVR=512
#define TOK   8192       // B*S tokens
#define HID_  2048
#define KVR_  512
#define D_    128

typedef __attribute__((ext_vector_type(16))) __bf16 v16bf;
typedef __attribute__((ext_vector_type(8)))  __bf16 v8bf;
typedef __attribute__((ext_vector_type(8)))  float  v8f;

// ---- WMMA fragment load (wave32, ISA 7.12.2 layout), bf16 source -----------
// Lane L holds row (L&15); lanes 0-15: K 0..7 & 16..23, lanes 16-31: K 8..15 & 24..31.
__device__ __forceinline__ v16bf load_frag(const __bf16* base, int ld, int lane) {
  const __bf16* p = base + (long)(lane & 15) * ld + ((lane & 16) ? 8 : 0);
  v8bf lo = *(const v8bf*)p;          // 16B-aligned (ld % 8 == 0, offset 0/16B)
  v8bf hi = *(const v8bf*)(p + 16);
  v16bf f;
#pragma unroll
  for (int i = 0; i < 8; ++i) { f[i] = lo[i]; f[i + 8] = hi[i]; }
  return f;
}

__device__ __forceinline__ v8f wmma_bf16(v16bf a, v16bf b, v8f c) {
  return __builtin_amdgcn_wmma_f32_16x16x32_bf16(
      /*neg_a=*/false, a, /*neg_b=*/false, b,
      /*c_mod=*/(short)0, c, /*reuse_a=*/false, /*reuse_b=*/false);
}

__device__ __forceinline__ void stval(float*  p, float v) { *p = v; }
__device__ __forceinline__ void stval(__bf16* p, float v) { *p = (__bf16)v; }

// ---- Tiled WMMA GEMM: C[M,N] = A[M,K] * B[N,K]^T + bias[N] -----------------
// Block = 128 threads (4 waves); block tile 128(M) x 64(N); wave tile 32x64.
// K-loop is ping-pong double-buffered in registers (all K multiples of 64).
template <typename TC>
__global__ __launch_bounds__(128)
void wmma_gemm_bias(const __bf16* __restrict__ A, const __bf16* __restrict__ B,
                    const float* __restrict__ bias, TC* __restrict__ C,
                    int K, int lda, int ldb, int ldc) {
  const int lane = threadIdx.x & 31;
  const int wave = threadIdx.x >> 5;
  const long m0 = (long)blockIdx.y * 128 + wave * 32;
  const long n0 = (long)blockIdx.x * 64;

  const __bf16* A0 = A + m0 * lda;
  const __bf16* A1 = A + (m0 + 16) * lda;
  const __bf16* Bp = B + n0 * ldb;

  v8f acc[8] = {};

  auto LOAD = [&](v16bf (&a)[2], v16bf (&b)[4], int k) {
    a[0] = load_frag(A0 + k, lda, lane);
    a[1] = load_frag(A1 + k, lda, lane);
#pragma unroll
    for (int j = 0; j < 4; ++j)
      b[j] = load_frag(Bp + (long)j * 16 * ldb + k, ldb, lane);
  };
  auto MMA = [&](v16bf (&a)[2], v16bf (&b)[4]) {
#pragma unroll
    for (int j = 0; j < 4; ++j) {
      acc[j]     = wmma_bf16(a[0], b[j], acc[j]);
      acc[4 + j] = wmma_bf16(a[1], b[j], acc[4 + j]);
    }
  };

  v16bf a0[2], b0[4], a1[2], b1[4];
  LOAD(a0, b0, 0);
  for (int k = 0; k < K; k += 64) {
    LOAD(a1, b1, k + 32);                       // loads overlap MMA(a0,b0)
    if (k + 128 < K) {                          // far prefetch -> global_prefetch_b8
      __builtin_prefetch(A0 + (long)(lane & 15) * lda + k + 128);
      __builtin_prefetch(Bp + (long)(lane & 15) * ldb + k + 128);
    }
    MMA(a0, b0);
    if (k + 64 < K) LOAD(a0, b0, k + 64);       // loads overlap MMA(a1,b1)
    MMA(a1, b1);
  }

  // C/D layout: VGPR r -> M = r + (lane>=16 ? 8 : 0); lane&15 -> N
  const int n  = lane & 15;
  const int mo = (lane & 16) ? 8 : 0;
#pragma unroll
  for (int j = 0; j < 4; ++j) {
    const long nn = n0 + j * 16 + n;
    const float bv = bias ? bias[nn] : 0.0f;
#pragma unroll
    for (int r = 0; r < 8; ++r) {
      stval(C + (m0      + mo + r) * (long)ldc + nn, acc[j][r]     + bv);
      stval(C + (m0 + 16 + mo + r) * (long)ldc + nn, acc[4 + j][r] + bv);
    }
  }
}

// ---- Streaming f32 -> bf16 convert (8 elems/thread, b128 in / b128 out) ----
__global__ __launch_bounds__(256)
void cvt_f32_bf16(const float* __restrict__ X, __bf16* __restrict__ Y, long n) {
  const long i = ((long)blockIdx.x * 256 + threadIdx.x) * 8;
  if (i >= n) return;
  float4 x0 = *(const float4*)(X + i);
  float4 x1 = *(const float4*)(X + i + 4);
  v8bf y;
  y[0] = (__bf16)x0.x; y[1] = (__bf16)x0.y; y[2] = (__bf16)x0.z; y[3] = (__bf16)x0.w;
  y[4] = (__bf16)x1.x; y[5] = (__bf16)x1.y; y[6] = (__bf16)x1.z; y[7] = (__bf16)x1.w;
  *(v8bf*)(Y + i) = y;
}

// ---- RMS over rows of 512 (f32 in, bf16 out): y = x*rsqrt(mean(x^2)+eps)*(1+w)
__global__ __launch_bounds__(256)
void rms_rows_512(const float* __restrict__ X, const float* __restrict__ w,
                  __bf16* __restrict__ Y) {
  const int lane = threadIdx.x & 31;
  const int wave = threadIdx.x >> 5;
  const long row = (long)blockIdx.x * 8 + wave;
  const float* x = X + row * KVR_;
  __bf16*      y = Y + row * KVR_;

  float v[16];
  float s = 0.0f;
#pragma unroll
  for (int i = 0; i < 16; ++i) { v[i] = x[i * 32 + lane]; s += v[i] * v[i]; }
#pragma unroll
  for (int off = 16; off; off >>= 1) s += __shfl_xor(s, off, 32);
  const float scale = rsqrtf(s * (1.0f / (float)KVR_) + 1e-6f);
#pragma unroll
  for (int i = 0; i < 16; ++i) {
    const int c = i * 32 + lane;
    y[c] = (__bf16)(v[i] * scale * (1.0f + w[c]));
  }
}

// ---- Head-fold of o_w: o_eff[i,d] = sum_h o_w[i, h*128+d]  (f32 -> bf16) ---
__global__ __launch_bounds__(256)
void reduce_ow(const float* __restrict__ ow, __bf16* __restrict__ oeff) {
  const long idx = (long)blockIdx.x * 256 + threadIdx.x;   // HID_*D_ threads
  const long i = idx >> 7;
  const int  d = (int)(idx & 127);
  float s = 0.0f;
#pragma unroll
  for (int h = 0; h < 16; ++h) s += ow[i * HID_ + h * D_ + d];
  oeff[idx] = (__bf16)s;
}

// ---------------------------------------------------------------------------
extern "C" void kernel_launch(void* const* d_in, const int* in_sizes, int n_in,
                              void* d_out, int out_size, void* d_ws, size_t ws_size,
                              hipStream_t stream) {
  (void)in_sizes; (void)n_in; (void)out_size; (void)ws_size;
  // setup_inputs() order. q-path / rope / mask / logits unused: the softmax is
  // over a size-1 key axis so p==1 identically, and out = v (per head) @ o_w^T.
  const float* hidden  = (const float*)d_in[0];
  const float* kv_a_w  = (const float*)d_in[11];
  const float* kv_a_b  = (const float*)d_in[12];
  const float* kv_norm = (const float*)d_in[13];
  const float* kv_b_w  = (const float*)d_in[14];
  const float* kv_b_b  = (const float*)d_in[15];
  const float* o_w     = (const float*)d_in[16];
  float*       out     = (float*)d_out;

  // Workspace layout (all 16B+ aligned)
  char*   ws       = (char*)d_ws;
  __bf16* hid_bf   = (__bf16*)(ws);                               // 32 MB
  __bf16* kva_bf   = (__bf16*)(ws + 33554432);                    //  2 MB
  __bf16* kvbv_bf  = (__bf16*)(ws + 35651584);                    // 128 KB
  __bf16* oeff     = (__bf16*)(ws + 35782656);                    // 512 KB
  float*  mid      = (float*) (ws + 36306944);                    // 16 MB
  __bf16* midn     = (__bf16*)(ws + 53084160);                    //  8 MB
  __bf16* vbuf     = (__bf16*)(ws + 61472768);                    //  2 MB

  const long n_hid = (long)TOK * HID_;          // 16,777,216
  const long n_kva = (long)KVR_ * HID_;         //  1,048,576
  const long n_kvb = (long)D_ * KVR_;           //     65,536 (v half only)

  // 0) precision staging: f32 -> bf16 for all GEMM operands
  cvt_f32_bf16<<<(unsigned)(n_hid / 2048), 256, 0, stream>>>(hidden, hid_bf, n_hid);
  cvt_f32_bf16<<<(unsigned)(n_kva / 2048), 256, 0, stream>>>(kv_a_w, kva_bf, n_kva);
  cvt_f32_bf16<<<(unsigned)(n_kvb / 2048), 256, 0, stream>>>(
      kv_b_w + (size_t)D_ * KVR_, kvbv_bf, n_kvb);
  reduce_ow<<<(HID_ * D_) / 256, 256, 0, stream>>>(o_w, oeff);

  // 1) mid = hidden @ kv_a_w^T + kv_a_b         [8192 x 512, K=2048] -> f32
  wmma_gemm_bias<float><<<dim3(KVR_ / 64, TOK / 128), 128, 0, stream>>>(
      hid_bf, kva_bf, kv_a_b, mid, HID_, HID_, HID_, KVR_);

  // 2) midn = RMS(mid) * (1 + kv_norm_w) -> bf16
  rms_rows_512<<<TOK / 8, 256, 0, stream>>>(mid, kv_norm, midn);

  // 3) v = midn @ kv_b_w[128:256]^T + kv_b_b[128:256]   [8192 x 128, K=512] -> bf16
  wmma_gemm_bias<__bf16><<<dim3(D_ / 64, TOK / 128), 128, 0, stream>>>(
      midn, kvbv_bf, kv_b_b + D_, vbuf, KVR_, KVR_, KVR_, D_);

  // 4) out = v @ o_eff^T                        [8192 x 2048, K=128] -> f32
  wmma_gemm_bias<float><<<dim3(HID_ / 64, TOK / 128), 128, 0, stream>>>(
      vbuf, oeff, nullptr, out, D_, D_, D_, HID_);
}
